// CircumpunctNodeV6_64828236366566
// MI455X (gfx1250) — matrix operand order repaired and checked by reading, not compiled, not run on hardware
//
#include <hip/hip_runtime.h>
#include <hip/hip_bf16.h>
#include <math.h>

// ---------------------------------------------------------------------------
// CircumpunctNodeV6 for MI455X (gfx1250, wave32, WMMA 16x16x32 f16/f32-acc)
// ---------------------------------------------------------------------------
#define B_  2
#define T_  1024
#define D_  1024
#define H_  16
#define DH_ 64
#define DV_ 32

typedef __attribute__((ext_vector_type(16))) _Float16 v16h;
typedef __attribute__((ext_vector_type(8)))  float    v8f;

__device__ __forceinline__ v8f zero8() { v8f z = {}; return z; }

__device__ __forceinline__ v8f wmma16(v16h a, v16h b, v8f c) {
  // D = A(16x32 f16) * B(32x16 f16) + C(16x16 f32)
  return __builtin_amdgcn_wmma_f32_16x16x32_f16(false, a, false, b, (short)0, c,
                                                false, false);
}

// A operand: row-major source, rows r0..r0+15, cols c0..c0+31.
// lane L: M = L&15, c = L>>4 ; halves i: col = 16c+i
__device__ __forceinline__ v16h loadA_rm(const _Float16* src, int ld, int r0, int c0) {
  const int lane = threadIdx.x & 31;
  const int M = lane & 15, c = lane >> 4;
  const _Float16* p = src + (size_t)(r0 + M) * ld + c0;
  v16h a;
#pragma unroll
  for (int i = 0; i < 8; ++i) a[i] = p[16 * c + i];
#pragma unroll
  for (int i = 0; i < 8; ++i) a[8 + i] = p[16 * c + 8 + i];
  return a;
}

// B operand from an (N x K) row-major source (weight (out,in) or K-matrix):
// B[k][n] = src[(n0+N)*ld + k0 + (16c+i)]
__device__ __forceinline__ v16h loadB_nk_h(const _Float16* src, int ld, int n0, int k0) {
  const int lane = threadIdx.x & 31;
  const int N = lane & 15, c = lane >> 4;
  const _Float16* p = src + (size_t)(n0 + N) * ld + k0 + 16 * c;
  v16h b;
#pragma unroll
  for (int i = 0; i < 16; ++i) b[i] = p[i];
  return b;
}

__device__ __forceinline__ v16h loadB_nk_f32(const float* src, int ld, int n0, int k0) {
  const int lane = threadIdx.x & 31;
  const int N = lane & 15, c = lane >> 4;
  const float* p = src + (size_t)(n0 + N) * ld + k0 + 16 * c;
  v16h b;
#pragma unroll
  for (int i = 0; i < 16; ++i) b[i] = (_Float16)p[i];
  return b;
}

// B operand from a (K x N) row-major source (e.g. V): B[k][n] = src[(k0+16c+i)*ld + n0+N]
__device__ __forceinline__ v16h loadB_kn_h(const _Float16* src, int ld, int k0, int n0) {
  const int lane = threadIdx.x & 31;
  const int N = lane & 15, c = lane >> 4;
  const _Float16* p = src + (size_t)(k0 + 16 * c) * ld + n0 + N;
  v16h b;
#pragma unroll
  for (int i = 0; i < 16; ++i) b[i] = p[(size_t)i * ld];
  return b;
}

// ---------------------------------------------------------------------------
// k_prep: vemb = vertex_emb + 0.1 * spectral_emb @ Wspec^T ; zero head |.| acc
// ---------------------------------------------------------------------------
__global__ void k_prep(const float* __restrict__ vertex_emb,
                       const float* __restrict__ Wspec,
                       const float* __restrict__ spectral,
                       float* __restrict__ vemb, float* __restrict__ headAbs) {
  if (threadIdx.x < H_) headAbs[threadIdx.x] = 0.f;
  for (int e = threadIdx.x; e < 64 * DV_; e += blockDim.x) {
    int x = e >> 5, v = e & 31;
    float a = 0.f;
#pragma unroll
    for (int g = 0; g < 6; ++g) a += spectral[x * 6 + g] * Wspec[v * 6 + g];
    vemb[e] = vertex_emb[e] + 0.1f * a;
  }
}

// ---------------------------------------------------------------------------
// k_qkv: per-head LN + WMMA Q/K/V projection (gate on Q,K), f16 out
// layout Qh/Kh/Vh: [(b*H + h)*T + t]*DH + d
// ---------------------------------------------------------------------------
__global__ void k_qkv(const float* __restrict__ x, const float* __restrict__ gate,
                      const float* __restrict__ cn_g, const float* __restrict__ cn_b,
                      const float* __restrict__ Wq, const float* __restrict__ Wk,
                      const float* __restrict__ Wv,
                      _Float16* __restrict__ Qh, _Float16* __restrict__ Kh,
                      _Float16* __restrict__ Vh) {
  const int widx = threadIdx.x >> 5;
  const int wave = blockIdx.x * (blockDim.x >> 5) + widx;
  const int qt = wave & 63, h = (wave >> 6) & 15, b = wave >> 10;
  const int lane = threadIdx.x & 31;
  __shared__ _Float16 hB[8][16 * DH_];
  __shared__ float gB[8][16];
  _Float16* hT = hB[widx];
  float* gT = gB[widx];

  // --- layer norm (2 lanes per token, shuffle-combined) ---
  {
    const int tok = lane & 15, part = lane >> 4;
    const int t = qt * 16 + tok;
    const float* xr = x + ((size_t)(b * T_ + t)) * D_ + h * DH_;
    float s = 0.f, s2 = 0.f;
#pragma unroll
    for (int j = 0; j < 32; ++j) {
      float v = xr[part * 32 + j];
      s += v; s2 += v * v;
    }
    s += __shfl_xor(s, 16); s2 += __shfl_xor(s2, 16);
    const float mu = s * (1.f / DH_);
    const float rstd = rsqrtf(s2 * (1.f / DH_) - mu * mu + 1e-5f);
#pragma unroll
    for (int j = 0; j < 32; ++j) {
      int col = part * 32 + j;
      float v = (xr[col] - mu) * rstd * cn_g[h * DH_ + col] + cn_b[h * DH_ + col];
      hT[tok * DH_ + col] = (_Float16)v;
    }
    if (part == 0) gT[tok] = gate[b * T_ + t];
  }

  const int N = lane & 15, sub = lane >> 4;
  const size_t obase = ((size_t)(b * H_ + h)) * T_ * DH_;
  const float* Wm[3] = {Wq + h * DH_ * DH_, Wk + h * DH_ * DH_, Wv + h * DH_ * DH_};
  _Float16* Om[3] = {Qh, Kh, Vh};
#pragma unroll
  for (int m = 0; m < 3; ++m) {
#pragma unroll
    for (int nt = 0; nt < 4; ++nt) {
      v8f acc = zero8();
      acc = wmma16(loadA_rm(hT, DH_, 0, 0),  loadB_nk_f32(Wm[m], DH_, nt * 16, 0),  acc);
      acc = wmma16(loadA_rm(hT, DH_, 0, 32), loadB_nk_f32(Wm[m], DH_, nt * 16, 32), acc);
#pragma unroll
      for (int r = 0; r < 8; ++r) {
        const int row = r + 8 * sub;
        const int t = qt * 16 + row;
        float v = acc[r];
        if (m < 2) v *= gT[row];          // gate on Q and K
        Om[m][obase + (size_t)t * DH_ + nt * 16 + N] = (_Float16)v;
      }
    }
  }
}

// ---------------------------------------------------------------------------
// k_attn: flash attention per (b,h,qtile): WMMA QK^T + online softmax + WMMA PV
// center layout: (b,t, h*64+d) f32 with residual xh added
// ---------------------------------------------------------------------------
__global__ void k_attn(const float* __restrict__ x, const _Float16* __restrict__ Qh,
                       const _Float16* __restrict__ Kh, const _Float16* __restrict__ Vh,
                       float* __restrict__ center) {
  const int widx = threadIdx.x >> 5;
  const int wave = blockIdx.x * (blockDim.x >> 5) + widx;
  const int qt = wave & 63, h = (wave >> 6) & 15, b = wave >> 10;
  const int lane = threadIdx.x & 31;
  const int N = lane & 15, sub = lane >> 4;
  __shared__ _Float16 pB[8][16 * 32];
  _Float16* pT = pB[widx];

  const size_t hb = ((size_t)(b * H_ + h)) * T_ * DH_;
  const _Float16* Qb = Qh + hb;
  const _Float16* Kb = Kh + hb;
  const _Float16* Vb = Vh + hb;

  const v16h aq0 = loadA_rm(Qb, DH_, qt * 16, 0);
  const v16h aq1 = loadA_rm(Qb, DH_, qt * 16, 32);

  v8f o[4]; float m[8], l[8];
#pragma unroll
  for (int i = 0; i < 4; ++i) o[i] = zero8();
#pragma unroll
  for (int r = 0; r < 8; ++r) { m[r] = -1e30f; l[r] = 0.f; }

  const int nChunks = (qt * 16 + 15) / 32 + 1;  // causal key chunks of 32
  for (int c = 0; c < nChunks; ++c) {
    const int key0 = c * 32;
    __builtin_prefetch(Kb + (size_t)(key0 + 32) * DH_, 0, 1);
    __builtin_prefetch(Vb + (size_t)(key0 + 32) * DH_, 0, 1);
    v8f s0 = zero8(), s1 = zero8();
    s0 = wmma16(aq0, loadB_nk_h(Kb, DH_, key0, 0),       s0);
    s0 = wmma16(aq1, loadB_nk_h(Kb, DH_, key0, 32),      s0);
    s1 = wmma16(aq0, loadB_nk_h(Kb, DH_, key0 + 16, 0),  s1);
    s1 = wmma16(aq1, loadB_nk_h(Kb, DH_, key0 + 16, 32), s1);

#pragma unroll
    for (int r = 0; r < 8; ++r) {
      const int row = qt * 16 + r + 8 * sub;
      const int c0 = key0 + N, c1 = key0 + 16 + N;
      float a0 = (c0 <= row) ? s0[r] * 0.125f : -1e30f;
      float a1 = (c1 <= row) ? s1[r] * 0.125f : -1e30f;
      float v = fmaxf(a0, a1);
      v = fmaxf(v, __shfl_xor(v, 1));
      v = fmaxf(v, __shfl_xor(v, 2));
      v = fmaxf(v, __shfl_xor(v, 4));
      v = fmaxf(v, __shfl_xor(v, 8));
      const float nm = fmaxf(m[r], v);
      const float f = __expf(m[r] - nm);
      m[r] = nm;
      const float p0 = __expf(a0 - nm);
      const float p1 = __expf(a1 - nm);
      float sum = p0 + p1;
      sum += __shfl_xor(sum, 1);
      sum += __shfl_xor(sum, 2);
      sum += __shfl_xor(sum, 4);
      sum += __shfl_xor(sum, 8);
      l[r] = l[r] * f + sum;
      o[0][r] *= f; o[1][r] *= f; o[2][r] *= f; o[3][r] *= f;
      pT[(r + 8 * sub) * 32 + N] = (_Float16)p0;
      pT[(r + 8 * sub) * 32 + 16 + N] = (_Float16)p1;
    }
    const v16h ap = loadA_rm(pT, 32, 0, 0);
#pragma unroll
    for (int nt = 0; nt < 4; ++nt)
      o[nt] = wmma16(ap, loadB_kn_h(Vb, DH_, key0, nt * 16), o[nt]);
  }

#pragma unroll
  for (int r = 0; r < 8; ++r) {
    const float inv = 1.f / l[r];
    const int t = qt * 16 + r + 8 * sub;
    const size_t base = ((size_t)(b * T_ + t)) * D_ + h * DH_;
#pragma unroll
    for (int nt = 0; nt < 4; ++nt) {
      const int col = nt * 16 + N;
      center[base + col] = o[nt][r] * inv + x[base + col];
    }
  }
}

// ---------------------------------------------------------------------------
// k_bound1: per-(b,t,h) boundary LN + gates + v_scores (pre-softmax) + |.| acc
// (inner loops capped unroll: <3% of FLOPs, keep I-cache footprint small)
// ---------------------------------------------------------------------------
__global__ void k_bound1(const float* __restrict__ x, const float* __restrict__ bn_g,
                         const float* __restrict__ bn_b, const float* __restrict__ Wgate,
                         const float* __restrict__ bgate, const float* __restrict__ Wg2v,
                         const float* __restrict__ nodeBiasTab, const int* __restrict__ nodeIdx,
                         const float* __restrict__ adj, const float* __restrict__ vemb,
                         float* __restrict__ vsc, float* __restrict__ headAbs) {
  const int idx = blockIdx.x * blockDim.x + threadIdx.x;
  if (idx >= B_ * T_ * H_) return;
  const int h = idx & 15, t = (idx >> 4) & (T_ - 1), b = idx >> 14;
  const float* xr = x + ((size_t)(b * T_ + t)) * D_ + h * DH_;

  float s = 0.f, s2 = 0.f;
#pragma unroll 8
  for (int j = 0; j < DH_; ++j) { float v = xr[j]; s += v; s2 += v * v; }
  const float mu = s * (1.f / DH_);
  const float rstd = rsqrtf(s2 * (1.f / DH_) - mu * mu + 1e-5f);
  float hb[DH_];
#pragma unroll 8
  for (int j = 0; j < DH_; ++j)
    hb[j] = (xr[j] - mu) * rstd * bn_g[h * DH_ + j] + bn_b[h * DH_ + j];

  const int ni = (nodeIdx[0] < 8) ? nodeIdx[0] : 8;
  float g6[6];
  for (int g = 0; g < 6; ++g) {
    float a = bgate[h * 6 + g] + nodeBiasTab[ni * 6 + g];
#pragma unroll 8
    for (int j = 0; j < DH_; ++j) a += Wgate[(h * 6 + g) * DH_ + j] * hb[j];
    g6[g] = 1.f / (1.f + __expf(-a));
  }
  float gp[DV_];
#pragma unroll 4
  for (int v = 0; v < DV_; ++v) {
    float a = 0.f;
#pragma unroll
    for (int g = 0; g < 6; ++g) a += Wg2v[(h * DV_ + v) * 6 + g] * g6[g];
    gp[v] = a;
  }
  float vs[64];
#pragma unroll 4
  for (int xx = 0; xx < 64; ++xx) {
    float a = 0.f;
#pragma unroll 8
    for (int v = 0; v < DV_; ++v) a += gp[v] * vemb[xx * DV_ + v];
    vs[xx] = a;
  }
  float absum = 0.f;
  float* vout = vsc + (size_t)idx * 64;
  for (int y = 0; y < 64; ++y) {
    float a = vs[y];
#pragma unroll 8
    for (int xx = 0; xx < 64; ++xx) a += 0.1f * vs[xx] * adj[xx * 64 + y];
    vout[y] = a;
    absum += fabsf(a);
  }
  atomicAdd(&headAbs[h], absum);
}

// ---------------------------------------------------------------------------
// k_bound2: softmax(vs/mean_abs*3), top-8 hard (STE fwd == hard), ctx, out
// ---------------------------------------------------------------------------
__global__ void k_bound2(const float* __restrict__ x, const float* __restrict__ vsc,
                         const float* __restrict__ headAbs, const float* __restrict__ vertex_emb,
                         const float* __restrict__ Wv2h, float* __restrict__ boundary) {
  const int idx = blockIdx.x * blockDim.x + threadIdx.x;
  if (idx >= B_ * T_ * H_) return;
  const int h = idx & 15, t = (idx >> 4) & (T_ - 1), b = idx >> 14;
  const float mean_abs = fmaxf(headAbs[h] * (1.f / (float)(B_ * T_ * 64)), 1e-6f);
  const float invm = 3.f / mean_abs;

  const float* vin = vsc + (size_t)idx * 64;
  float z[64];
  float mx = -1e30f;
#pragma unroll 8
  for (int y = 0; y < 64; ++y) { z[y] = vin[y] * invm; mx = fmaxf(mx, z[y]); }
  float sum = 0.f;
#pragma unroll 8
  for (int y = 0; y < 64; ++y) { z[y] = __expf(z[y] - mx); sum += z[y]; }
  const float inv = 1.f / sum;
#pragma unroll 8
  for (int y = 0; y < 64; ++y) z[y] *= inv;  // soft_pi

  // threshold = 8th largest
  float tmp[64];
#pragma unroll 8
  for (int y = 0; y < 64; ++y) tmp[y] = z[y];
  float thr = 0.f;
  for (int it = 0; it < 8; ++it) {
    int bi = 0; float bv = -1.f;
    for (int y = 0; y < 64; ++y) if (tmp[y] > bv) { bv = tmp[y]; bi = y; }
    thr = bv; tmp[bi] = -2.f;
  }
  float hsum = 0.f;
#pragma unroll 8
  for (int y = 0; y < 64; ++y) { z[y] = (z[y] >= thr) ? z[y] : 0.f; hsum += z[y]; }
  const float hn = 1.f / (hsum + 1e-8f);

  float ctx[DV_];
#pragma unroll 4
  for (int v = 0; v < DV_; ++v) {
    float a = 0.f;
#pragma unroll 8
    for (int y = 0; y < 64; ++y) a += z[y] * hn * vertex_emb[y * DV_ + v];
    ctx[v] = a;
  }
  const size_t base = ((size_t)(b * T_ + t)) * D_ + h * DH_;
#pragma unroll 4
  for (int d = 0; d < DH_; ++d) {
    float a = 0.f;
#pragma unroll 8
    for (int v = 0; v < DV_; ++v) a += Wv2h[(h * DH_ + d) * DV_ + v] * ctx[v];
    boundary[base + d] = x[base + d] + a;
  }
}

// ---------------------------------------------------------------------------
// k_field: cat(center,boundary)->Wcomb (WMMA), LN, Wup+GELU (WMMA), Wdn (WMMA)
// emits merged field as f16 (B,T,D) for the output GEMM
// ---------------------------------------------------------------------------
__global__ void k_field(const float* __restrict__ center, const float* __restrict__ boundary,
                        const float* __restrict__ Wcomb, const float* __restrict__ bcomb,
                        const float* __restrict__ pn_g, const float* __restrict__ pn_b,
                        const float* __restrict__ Wup, const float* __restrict__ bup,
                        const float* __restrict__ Wdn, const float* __restrict__ bdn,
                        _Float16* __restrict__ mergedH) {
  const int widx = threadIdx.x >> 5;
  const int wave = blockIdx.x * (blockDim.x >> 5) + widx;
  const int qt = wave & 63, h = (wave >> 6) & 15, b = wave >> 10;
  const int lane = threadIdx.x & 31;
  const int N = lane & 15, sub = lane >> 4;
  __shared__ _Float16 catB[8][16 * 128];   // reused as up-activation buffer
  __shared__ _Float16 phB[8][16 * 64];
  _Float16* catT = catB[widx];
  _Float16* phT = phB[widx];

  // stage cat = [center, boundary] as f16
  for (int e = lane; e < 16 * 128; e += 32) {
    const int row = e >> 7, col = e & 127;
    const int t = qt * 16 + row;
    const size_t base = ((size_t)(b * T_ + t)) * D_ + h * DH_;
    const float v = (col < 64) ? center[base + col] : boundary[base + col - 64];
    catT[e] = (_Float16)v;
  }

  const float* Wc = Wcomb + h * DH_ * 128;
  v8f phi[4];
#pragma unroll
  for (int nt = 0; nt < 4; ++nt) {
    v8f acc = zero8();
#pragma unroll
    for (int k0 = 0; k0 < 128; k0 += 32)
      acc = wmma16(loadA_rm(catT, 128, 0, k0), loadB_nk_f32(Wc, 128, nt * 16, k0), acc);
#pragma unroll
    for (int r = 0; r < 8; ++r) acc[r] += bcomb[h * DH_ + nt * 16 + N];
    phi[nt] = acc;
  }

  // per-row LN of phi (rows live across 16-lane groups)
  float mu[8], rs[8];
#pragma unroll
  for (int r = 0; r < 8; ++r) {
    float s = phi[0][r] + phi[1][r] + phi[2][r] + phi[3][r];
    float q = phi[0][r] * phi[0][r] + phi[1][r] * phi[1][r] +
              phi[2][r] * phi[2][r] + phi[3][r] * phi[3][r];
    s += __shfl_xor(s, 1); s += __shfl_xor(s, 2); s += __shfl_xor(s, 4); s += __shfl_xor(s, 8);
    q += __shfl_xor(q, 1); q += __shfl_xor(q, 2); q += __shfl_xor(q, 4); q += __shfl_xor(q, 8);
    mu[r] = s * (1.f / DH_);
    rs[r] = rsqrtf(q * (1.f / DH_) - mu[r] * mu[r] + 1e-5f);
  }
#pragma unroll
  for (int nt = 0; nt < 4; ++nt) {
#pragma unroll
    for (int r = 0; r < 8; ++r) {
      const int col = nt * 16 + N;
      const float v = (phi[nt][r] - mu[r]) * rs[r] * pn_g[h * DH_ + col] + pn_b[h * DH_ + col];
      phT[(r + 8 * sub) * DH_ + col] = (_Float16)v;
    }
  }

  // up projection + exact GELU -> reuse catT as (16 x 128) activation buffer
  const float* Wu = Wup + h * 128 * DH_;
#pragma unroll
  for (int nt = 0; nt < 8; ++nt) {
    v8f acc = zero8();
    acc = wmma16(loadA_rm(phT, DH_, 0, 0),  loadB_nk_f32(Wu, DH_, nt * 16, 0),  acc);
    acc = wmma16(loadA_rm(phT, DH_, 0, 32), loadB_nk_f32(Wu, DH_, nt * 16, 32), acc);
#pragma unroll
    for (int r = 0; r < 8; ++r) {
      const int col = nt * 16 + N;
      float v = acc[r] + bup[h * 128 + col];
      v = 0.5f * v * (1.f + erff(v * 0.70710678f));
      catT[(r + 8 * sub) * 128 + col] = (_Float16)v;
    }
  }

  // down projection + residual phi_in -> merged f16
  const float* Wd = Wdn + h * DH_ * 128;
#pragma unroll
  for (int nt = 0; nt < 4; ++nt) {
    v8f acc = zero8();
#pragma unroll
    for (int k0 = 0; k0 < 128; k0 += 32)
      acc = wmma16(loadA_rm(catT, 128, 0, k0), loadB_nk_f32(Wd, 128, nt * 16, k0), acc);
#pragma unroll
    for (int r = 0; r < 8; ++r) {
      const int col = nt * 16 + N;
      const int t = qt * 16 + r + 8 * sub;
      const float v = acc[r] + bdn[h * DH_ + col] + phi[nt][r];
      mergedH[((size_t)(b * T_ + t)) * D_ + h * DH_ + col] = (_Float16)v;
    }
  }
}

// ---------------------------------------------------------------------------
// k_wcvt: Wout f32 -> f16
// ---------------------------------------------------------------------------
__global__ void k_wcvt(const float* __restrict__ W, _Float16* __restrict__ Wh, int n) {
  const int i = blockIdx.x * blockDim.x + threadIdx.x;
  if (i < n) Wh[i] = (_Float16)W[i];
}

// ---------------------------------------------------------------------------
// k_gemm: proj = merged(2048x1024 f16) @ Wout^T(1024x1024) + bout, WMMA tiled.
// All 8 waves in a block share the same 16-row A tile -> stage each 16x32 f16
// A chunk into LDS once per block with gfx1250 async DMA
// (global_load_async_to_lds_b128, ASYNCcnt), then every wave reads it from
// LDS for its own 64 N columns. 8x reduction of A global traffic.
// ---------------------------------------------------------------------------
__global__ void k_gemm(const _Float16* __restrict__ A, const _Float16* __restrict__ Wh,
                       const float* __restrict__ bout, float* __restrict__ proj) {
  const int widx = threadIdx.x >> 5;
  const int mt = blockIdx.x >> 1;                    // 128 M tiles
  const int n0 = (blockIdx.x & 1) * 512 + widx * 64; // 2 x 8 waves x 64 N cols
  const int lane = threadIdx.x & 31;
  const int N = lane & 15, sub = lane >> 4;
  __shared__ _Float16 aT[16 * 32];

  v8f acc[4];
#pragma unroll
  for (int i = 0; i < 4; ++i) acc[i] = zero8();

  const unsigned ldsBase = (unsigned)(uintptr_t)(&aT[0]);
  const int row = threadIdx.x >> 2, seg = threadIdx.x & 3;  // threads 0..63 stage
  const unsigned loff = ldsBase + (unsigned)((row * 32 + seg * 8) * sizeof(_Float16));

  for (int k0 = 0; k0 < D_; k0 += 32) {
    if (threadIdx.x < 64) {
      const _Float16* gp = A + (size_t)(mt * 16 + row) * D_ + k0 + seg * 8;
      // async DMA: 16B per lane, LDS dest offset in VGPR, tracked by ASYNCcnt
      asm volatile("global_load_async_to_lds_b128 %0, %1, off"
                   :: "v"(loff), "v"((unsigned long long)(uintptr_t)gp)
                   : "memory");
      asm volatile("s_wait_asynccnt 0x0" ::: "memory");
    }
    __syncthreads();
    __builtin_prefetch(Wh + (size_t)(n0 + N) * D_ + k0 + 64, 0, 1);
    const v16h a = loadA_rm(aT, 32, 0, 0);
#pragma unroll
    for (int j = 0; j < 4; ++j)
      acc[j] = wmma16(a, loadB_nk_h(Wh, D_, n0 + j * 16, k0), acc[j]);
    __syncthreads();   // protect aT before next chunk overwrites it
  }
#pragma unroll
  for (int j = 0; j < 4; ++j) {
#pragma unroll
    for (int r = 0; r < 8; ++r) {
      const int m = mt * 16 + r + 8 * sub;
      const int col = n0 + j * 16 + N;
      proj[(size_t)m * D_ + col] = acc[j][r] + bout[col];
    }
  }
}

// ---------------------------------------------------------------------------
// k_final_ln: out = LN(x + proj) over D=1024, block per token
// ---------------------------------------------------------------------------
__global__ void k_final_ln(const float* __restrict__ x, const float* __restrict__ proj,
                           const float* __restrict__ g, const float* __restrict__ bb,
                           float* __restrict__ out) {
  const int tok = blockIdx.x;
  const float* xr = x + (size_t)tok * D_;
  const float* pr = proj + (size_t)tok * D_;
  __shared__ float red[16];
  float y[4];
  float s = 0.f, s2 = 0.f;
#pragma unroll
  for (int i = 0; i < 4; ++i) {
    const int col = threadIdx.x + i * 256;
    y[i] = xr[col] + pr[col];
    s += y[i]; s2 += y[i] * y[i];
  }
#pragma unroll
  for (int msk = 1; msk < 32; msk <<= 1) { s += __shfl_xor(s, msk); s2 += __shfl_xor(s2, msk); }
  const int w = threadIdx.x >> 5;
  if ((threadIdx.x & 31) == 0) { red[w] = s; red[8 + w] = s2; }
  __syncthreads();
  float ts = 0.f, ts2 = 0.f;
#pragma unroll
  for (int i = 0; i < 8; ++i) { ts += red[i]; ts2 += red[8 + i]; }
  const float mu = ts * (1.f / D_);
  const float rstd = rsqrtf(ts2 * (1.f / D_) - mu * mu + 1e-5f);
  float* orow = out + (size_t)tok * D_;
#pragma unroll
  for (int i = 0; i < 4; ++i) {
    const int col = threadIdx.x + i * 256;
    orow[col] = (y[i] - mu) * rstd * g[col] + bb[col];
  }
}

// ---------------------------------------------------------------------------
extern "C" void kernel_launch(void* const* d_in, const int* in_sizes, int n_in,
                              void* d_out, int out_size, void* d_ws, size_t ws_size,
                              hipStream_t stream) {
  (void)in_sizes; (void)n_in; (void)out_size; (void)ws_size;
  const float* x        = (const float*)d_in[0];
  const float* gatep    = (const float*)d_in[1];
  const int*   node_idx = (const int*)  d_in[3];
  const float* cn_g     = (const float*)d_in[4];
  const float* cn_b     = (const float*)d_in[5];
  const float* Wq       = (const float*)d_in[6];
  const float* Wk       = (const float*)d_in[7];
  const float* Wv       = (const float*)d_in[8];
  const float* bn_g     = (const float*)d_in[9];
  const float* bn_b     = (const float*)d_in[10];
  const float* Wgate    = (const float*)d_in[11];
  const float* bgate    = (const float*)d_in[12];
  const float* Wg2v     = (const float*)d_in[13];
  const float* Wv2h     = (const float*)d_in[14];
  const float* Wcomb    = (const float*)d_in[15];
  const float* bcomb    = (const float*)d_in[16];
  const float* pn_g     = (const float*)d_in[17];
  const float* pn_b     = (const float*)d_in[18];
  const float* Wup      = (const float*)d_in[19];
  const float* bup      = (const float*)d_in[20];
  const float* Wdn      = (const float*)d_in[21];
  const float* bdn      = (const float*)d_in[22];
  const float* vertex   = (const float*)d_in[23];
  const float* Wspec    = (const float*)d_in[24];
  const float* nbias    = (const float*)d_in[25];
  const float* adj      = (const float*)d_in[26];
  const float* spectral = (const float*)d_in[27];
  const float* Wout     = (const float*)d_in[28];
  const float* bout     = (const float*)d_in[29];
  const float* ln_g     = (const float*)d_in[30];
  const float* ln_b     = (const float*)d_in[31];
  float* out = (float*)d_out;

  // workspace carve-out (~50 MB)
  char* w = (char*)d_ws;
  size_t off = 0;
  auto take = [&](size_t bytes) -> void* {
    void* p = w + off;
    off += (bytes + 255) & ~(size_t)255;
    return p;
  };
  const size_t nTok = (size_t)B_ * T_;
  _Float16* Qh      = (_Float16*)take(nTok * D_ * sizeof(_Float16));
  _Float16* Kh      = (_Float16*)take(nTok * D_ * sizeof(_Float16));
  _Float16* Vh      = (_Float16*)take(nTok * D_ * sizeof(_Float16));
  float*    center  = (float*)   take(nTok * D_ * sizeof(float));
  float*    boundary= (float*)   take(nTok * D_ * sizeof(float));
  float*    vsc     = (float*)   take(nTok * H_ * 64 * sizeof(float));
  float*    vemb    = (float*)   take(64 * DV_ * sizeof(float));
  float*    headAbs = (float*)   take(H_ * sizeof(float));
  _Float16* mergedH = (_Float16*)take(nTok * D_ * sizeof(_Float16));
  _Float16* WoutH   = (_Float16*)take((size_t)D_ * D_ * sizeof(_Float16));
  float*    proj    = (float*)   take(nTok * D_ * sizeof(float));

  const dim3 blk(256);
  k_prep<<<1, blk, 0, stream>>>(vertex, Wspec, spectral, vemb, headAbs);
  k_qkv<<<256, blk, 0, stream>>>(x, gatep, cn_g, cn_b, Wq, Wk, Wv, Qh, Kh, Vh);
  k_attn<<<256, blk, 0, stream>>>(x, Qh, Kh, Vh, center);
  k_bound1<<<(B_ * T_ * H_) / 256, blk, 0, stream>>>(x, bn_g, bn_b, Wgate, bgate, Wg2v,
                                                     nbias, node_idx, adj, vemb, vsc, headAbs);
  k_bound2<<<(B_ * T_ * H_) / 256, blk, 0, stream>>>(x, vsc, headAbs, vertex, Wv2h, boundary);
  k_field<<<256, blk, 0, stream>>>(center, boundary, Wcomb, bcomb, pn_g, pn_b,
                                   Wup, bup, Wdn, bdn, mergedH);
  k_wcvt<<<(D_ * D_) / 256, blk, 0, stream>>>(Wout, WoutH, D_ * D_);
  k_gemm<<<256, blk, 0, stream>>>(mergedH, WoutH, bout, proj);
  k_final_ln<<<(int)nTok, blk, 0, stream>>>(x, proj, ln_g, ln_b, out);
}
